// ActivateAttentionLayer_41601053229528
// MI455X (gfx1250) — matrix operand order repaired and checked by reading, not compiled, not run on hardware
//
#include <hip/hip_runtime.h>
#include <hip/hip_bf16.h>
#include <math.h>

// ---------------------------------------------------------------- constants
#define BB   256
#define NN   75
#define HH   6
#define OUTD 128
#define DD   768              // HH*OUTD
#define EMB  100
#define ROWS (BB*NN)          // 19200 rows, = 1200 * 16
#define MTILES (ROWS/16)      // 1200

// LDS B-panel chunk: 128 cols x 128 K bf16, padded row stride 272B (256B data + 16B pad)
#define CHUNK_K     128
#define LDS_ROW     272
#define CHUNK_BYTES (128 * LDS_ROW)     // 34816
#define NCHUNK      (DD / CHUNK_K)      // 6

typedef __attribute__((ext_vector_type(16))) __bf16 v16bf;
typedef __attribute__((ext_vector_type(8)))  __bf16 v8bf;
typedef __attribute__((ext_vector_type(8)))  float  v8f;
typedef __attribute__((ext_vector_type(4)))  unsigned int v4u;
typedef __attribute__((ext_vector_type(8)))  int v8i;
typedef __attribute__((ext_vector_type(4)))  int v4i;

#if __has_builtin(__builtin_amdgcn_tensor_load_to_lds) && __has_builtin(__builtin_amdgcn_s_wait_tensorcnt)
#define USE_TDM 1
#else
#define USE_TDM 0
#endif

// f32 -> bf16 round-to-nearest-even
__device__ __forceinline__ __bf16 f2bf(float f) {
    unsigned u = __builtin_bit_cast(unsigned, f);
    unsigned r = (u + 0x7FFFu + ((u >> 16) & 1u)) >> 16;
    unsigned short s = (unsigned short)r;
    return __builtin_bit_cast(__bf16, s);
}

__device__ __forceinline__ v8f wmma_bf16(v16bf a, v16bf b, v8f c) {
    return __builtin_amdgcn_wmma_f32_16x16x32_bf16(
        /*neg_a=*/false, a, /*neg_b=*/false, b,
        /*c_mod=*/(short)0, c, /*reuse_a=*/false, /*reuse_b=*/false);
}

// A fragment: 16x32 bf16, M = lane&15.
// lane<16 : v0..3 = K[ks..ks+7],  v4..7 = K[ks+16..ks+23], ks = kb*32
// lane>=16: same with ks = kb*32+8  (ISA 16-bit A layout)
__device__ __forceinline__ v16bf load_a_frag(const __bf16* A, int row, int kb, int lane) {
    int ks = kb * 32 + ((lane & 16) ? 8 : 0);
    const __bf16* p = A + (size_t)row * DD + ks;
    v8bf lo = *(const v8bf*)(p);
    v8bf hi = *(const v8bf*)(p + 16);
    v16bf r;
#pragma unroll
    for (int i = 0; i < 8; ++i) { r[i] = lo[i]; r[i + 8] = hi[i]; }
    return r;
}

// B fragment from LDS chunk buffer: col-major-in-K rows of 128 K elements, stride LDS_ROW.
// lane<16 holds K[kk..kk+7],K[kk+16..kk+23]; lane>=16 same with kk+8.
__device__ __forceinline__ v16bf load_b_lds(const char* buf, int col, int kkElem) {
    const __bf16* p = (const __bf16*)(buf + col * LDS_ROW + kkElem * 2);
    v8bf lo = *(const v8bf*)(p);
    v8bf hi = *(const v8bf*)(p + 16);
    v16bf r;
#pragma unroll
    for (int i = 0; i < 8; ++i) { r[i] = lo[i]; r[i + 8] = hi[i]; }
    return r;
}

#if USE_TDM
// Issue a TDM 2D tile load: 128 cols x 128 K bf16 from [col][k] global (stride DD)
// into LDS with 16B pad every 256B (row stride LDS_ROW). Wave-level op, EXEC ignored.
__device__ __forceinline__ void tdm_load_panel(unsigned ldsAddr, const void* gsrc) {
    unsigned long long ga = (unsigned long long)(uintptr_t)gsrc;
    v4u g0 = { 1u,                                        // count=1, user mode
               ldsAddr,                                   // lds_addr
               (unsigned)(ga & 0xFFFFFFFFull),            // global_addr[31:0]
               (unsigned)(((ga >> 32) & 0x01FFFFFFull) | (2u << 30)) }; // ga[56:32] | type=2
    v8i g1 = { (int)((1u << 16)      // data_size = 2 bytes
                   | (1u << 20)      // pad_enable
                   | (5u << 22)      // pad_interval: 64 DWORDs = 256B
                   | (3u << 25)),    // pad_amount: 4 DWORDs = 16B
               (int)((unsigned)DD << 16),   // tensor_dim0[15:0] in bits[63:48]
               (int)(128u << 16),           // tensor_dim1[15:0] in bits[95:80]
               (int)(128u << 16),           // tile_dim0 in bits[127:112]
               (int)128,                    // tile_dim1 in bits[143:128]
               (int)DD,                     // tensor_dim0_stride[31:0]
               0, 0 };
    v4i gz = { 0, 0, 0, 0 };
#if defined(__clang_major__) && (__clang_major__ >= 23)
    v8i gz8 = { 0, 0, 0, 0, 0, 0, 0, 0 };
    __builtin_amdgcn_tensor_load_to_lds(g0, g1, gz, gz, gz8, 0);
#else
    __builtin_amdgcn_tensor_load_to_lds(g0, g1, gz, gz, 0);
#endif
}
#endif

// ---------------------------------------------------------------- converters
__global__ void cvt_f32_bf16_kernel(const float* __restrict__ in, __bf16* __restrict__ out, int n) {
    int i = blockIdx.x * blockDim.x + threadIdx.x;
    if (i < n) out[i] = f2bf(in[i]);
}

// Wt[w][h][o][d] = W_w[h][d][o]   (3*6*128*768 elements)
__global__ void cvt_weights_kernel(const float* __restrict__ W1, const float* __restrict__ W2,
                                   const float* __restrict__ W3, __bf16* __restrict__ Wt) {
    int i = blockIdx.x * blockDim.x + threadIdx.x;
    if (i >= 3 * HH * OUTD * DD) return;
    int d = i % DD; int r = i / DD;
    int o = r % OUTD; r /= OUTD;
    int h = r % HH;  int w = r / HH;
    const float* W = (w == 0) ? W1 : (w == 1) ? W2 : W3;
    Wt[i] = f2bf(W[((size_t)h * DD + d) * OUTD + o]);
}

// ---------------------------------------------------------------- GEMM core (per-wave 16x128 tile)
// 8 waves per block share the LDS-staged B panel; chunked 128-K, TDM double-buffered when available.
__device__ __forceinline__ void gemm_tile_compute(const __bf16* __restrict__ featB,
                                                  const __bf16* __restrict__ Bt,  // [128 cols][DD] bf16
                                                  char* smem, int tid, int lane, int arow,
                                                  v8f acc[8]) {
    char* buf0 = smem;
    char* buf1 = smem + CHUNK_BYTES;
#if USE_TDM
    if (tid < 32) tdm_load_panel((unsigned)(uintptr_t)buf0, Bt);
#endif
    for (int c = 0; c < NCHUNK; ++c) {
        char* cur = (c & 1) ? buf1 : buf0;
#if USE_TDM
        if (tid < 32) {
            if (c + 1 < NCHUNK) {
                char* nxt = ((c + 1) & 1) ? buf1 : buf0;
                tdm_load_panel((unsigned)(uintptr_t)nxt, Bt + (size_t)(c + 1) * CHUNK_K);
                __builtin_amdgcn_s_wait_tensorcnt(1);
            } else {
                __builtin_amdgcn_s_wait_tensorcnt(0);
            }
        }
#else
        {   // cooperative copy fallback: 256 threads, 2 threads per column (64 elements each)
            int col  = tid >> 1;
            int half = tid & 1;
            const __bf16* src = Bt + (size_t)col * DD + c * CHUNK_K + half * 64;
            __bf16* dst = (__bf16*)(cur + col * LDS_ROW) + half * 64;
#pragma unroll
            for (int q = 0; q < 8; ++q)
                *(v8bf*)(dst + q * 8) = *(const v8bf*)(src + q * 8);
        }
#endif
        __syncthreads();
#pragma unroll
        for (int ks = 0; ks < 4; ++ks) {
            int kb = c * 4 + ks;
            __builtin_prefetch(featB + (size_t)arow * DD + (kb + 1) * 32, 0, 1);
            v16bf a = load_a_frag(featB, arow, kb, lane);
            int kkElem = ks * 32 + ((lane & 16) ? 8 : 0);
#pragma unroll
            for (int nt = 0; nt < 8; ++nt) {
                v16bf b = load_b_lds(cur, nt * 16 + (lane & 15), kkElem);
                acc[nt] = wmma_bf16(a, b, acc[nt]);
            }
        }
        __syncthreads();
    }
}

// grid (150, 18) block 256 (8 waves). blockIdx.y -> (w, h). Each wave: 16 rows x 128 cols.
__global__ void gemm_h_kernel(const __bf16* __restrict__ featB, const __bf16* __restrict__ WtB,
                              float* __restrict__ hbuf) {
    extern __shared__ char smem[];
    int tid = threadIdx.x;
    int wave = tid >> 5;
    int lane = tid & 31;
    int mt = blockIdx.x * 8 + wave;                // 0..1199
    int w  = blockIdx.y / HH;
    int h  = blockIdx.y % HH;
    const __bf16* Bt = WtB + (size_t)(w * HH + h) * OUTD * DD;

    v8f acc[8];
#pragma unroll
    for (int i = 0; i < 8; ++i) acc[i] = (v8f){0.f,0.f,0.f,0.f,0.f,0.f,0.f,0.f};

    int M0 = mt * 16;
    int arow = M0 + (lane & 15);
    gemm_tile_compute(featB, Bt, smem, tid, lane, arow, acc);

    // store: hbuf[w][(b*6+h)*75+n][o]
    float* out = hbuf + (size_t)w * ((size_t)BB * HH * NN * OUTD);
    int rbase = M0 + ((lane & 16) ? 8 : 0);
    int col0  = lane & 15;
#pragma unroll
    for (int nt = 0; nt < 8; ++nt) {
#pragma unroll
        for (int r = 0; r < 8; ++r) {
            int row = rbase + r;
            int bb = row / NN, nn = row % NN;
            out[((size_t)(bb * HH + h) * NN + nn) * OUTD + nt * 16 + col0] = acc[nt][r];
        }
    }
}

// gate_pre[i][o] = sum_d feat[i][d] * H_w[o][d].  grid (150, 6) block 256.
__global__ void gemm_gate_kernel(const __bf16* __restrict__ featB, const __bf16* __restrict__ HwB,
                                 float* __restrict__ gate_pre) {
    extern __shared__ char smem[];
    int tid = threadIdx.x;
    int wave = tid >> 5;
    int lane = tid & 31;
    int mt = blockIdx.x * 8 + wave;
    int ng = blockIdx.y;                            // column group of 128
    const __bf16* Bt = HwB + (size_t)ng * 128 * DD;

    v8f acc[8];
#pragma unroll
    for (int i = 0; i < 8; ++i) acc[i] = (v8f){0.f,0.f,0.f,0.f,0.f,0.f,0.f,0.f};

    int M0 = mt * 16;
    int arow = M0 + (lane & 15);
    gemm_tile_compute(featB, Bt, smem, tid, lane, arow, acc);

    int rbase = M0 + ((lane & 16) ? 8 : 0);
    int col0  = lane & 15;
#pragma unroll
    for (int nt = 0; nt < 8; ++nt) {
#pragma unroll
        for (int r = 0; r < 8; ++r) {
            int row = rbase + r;
            gate_pre[(size_t)row * DD + ng * 128 + nt * 16 + col0] = acc[nt][r];
        }
    }
}

// ---------------------------------------------------------------- argsort + embedding attention terms
__global__ void emb_kernel(const float* __restrict__ adj, const float* __restrict__ output1,
                           const float* __restrict__ output2, const float* __restrict__ embedding,
                           const float* __restrict__ w_src_o, const float* __restrict__ w_dst_o,
                           float* __restrict__ o1, float* __restrict__ o2) {
    int b = blockIdx.x;
    int tid = threadIdx.x;
    __shared__ float sv1[NN], sv2[NN];
    __shared__ int sArg1[NN], sArg2[NN], sMs[NN];
    if (tid < NN) {
        sv1[tid] = output1[b * NN + tid];
        sv2[tid] = output2[b * NN + tid];
    }
    __syncthreads();
    if (tid < NN) {
        float x1 = sv1[tid], x2 = sv2[tid];
        int r1 = 0, r2 = 0;
        for (int k = 0; k < NN; ++k) {
            float y1 = sv1[k], y2 = sv2[k];
            r1 += (y1 < x1) || (y1 == x1 && k < tid);
            r2 += (y2 < x2) || (y2 == x2 && k < tid);
        }
        sArg1[r1] = tid;
        sArg2[r2] = tid;
        float ssum = 0.f;
        const float* ar = adj + ((size_t)b * NN + tid) * NN;
        for (int m = 0; m < NN; ++m) ssum += ar[m];
        sMs[tid] = (ssum > 0.f) ? 1 : 0;
    }
    __syncthreads();
    for (int t = tid; t < NN * HH; t += blockDim.x) {
        int j = t % NN, h = t / NN;
        int i1 = (sArg1[j] + 1) * sMs[j];
        int i2 = (sArg2[j] + 1) * sMs[j];
        float v1 = 0.f, v2 = 0.f;
        if (i1 != 0) {
            const float* e = embedding + (size_t)i1 * EMB;
            const float* w = w_src_o + (size_t)h * EMB;
            for (int k = 0; k < EMB; ++k) v1 += tanhf(e[k]) * w[k];
        }
        if (i2 != 0) {
            const float* e = embedding + (size_t)i2 * EMB;
            const float* w = w_dst_o + (size_t)h * EMB;
            for (int k = 0; k < EMB; ++k) v2 += tanhf(e[k]) * w[k];
        }
        o1[((size_t)b * HH + h) * NN + j] = v1;
        o2[((size_t)b * HH + h) * NN + j] = v2;
    }
}

// ---------------------------------------------------------------- a_src / a_dst from tanh(h1)
__global__ void asrc_kernel(const float* __restrict__ h1, const float* __restrict__ w_src,
                            const float* __restrict__ w_dst, float* __restrict__ a_src,
                            float* __restrict__ a_dst) {
    int i = blockIdx.x * blockDim.x + threadIdx.x;   // over B*H*N
    if (i >= BB * HH * NN) return;
    int h = (i / NN) % HH;
    const float* row = h1 + (size_t)i * OUTD;
    const float* ws = w_src + h * OUTD;
    const float* wd = w_dst + h * OUTD;
    float s = 0.f, d = 0.f;
    for (int o = 0; o < OUTD; ++o) {
        float t = tanhf(row[o]);
        s += t * ws[o];
        d += t * wd[o];
    }
    a_src[i] = s;
    a_dst[i] = d;
}

// ---------------------------------------------------------------- masked double-softmax + aggregation
// One block per (b,h), 128 threads. Dynamic LDS: attn2, attn3, h2 tile, h3 tile.
// Writes elu(feat_out) into d_out (B,N,D).
__global__ void attn_kernel(const float* __restrict__ adj, const float* __restrict__ s_mask,
                            const float* __restrict__ a_src, const float* __restrict__ a_dst,
                            const float* __restrict__ o1, const float* __restrict__ o2,
                            const float* __restrict__ hbuf, const float* __restrict__ bias,
                            float* __restrict__ out) {
    extern __shared__ float dsm[];
    float* sA2 = dsm;                         // NN*(NN+1)
    float* sA3 = sA2 + NN * (NN + 1);         // NN*(NN+1)
    float* h2s = sA3 + NN * (NN + 1);         // NN*OUTD
    float* h3s = h2s + NN * OUTD;             // NN*OUTD

    int bh = blockIdx.x;
    int b = bh / HH, h = bh % HH;
    int tid = threadIdx.x;
    __shared__ float sSrc[NN], sDst[NN], sO1[NN], sO2[NN], sDiag[NN];

    const size_t HSZ = (size_t)BB * HH * NN * OUTD;
    const float* h1 = hbuf + (size_t)bh * NN * OUTD;
    const float* h2 = h1 + HSZ;
    const float* h3 = h1 + 2 * HSZ;

    // stage h2/h3 tiles (NN*OUTD floats each) via float4
    for (int t = tid; t < NN * OUTD / 4; t += blockDim.x) {
        ((float4*)h2s)[t] = ((const float4*)h2)[t];
        ((float4*)h3s)[t] = ((const float4*)h3)[t];
    }
    if (tid < NN) {
        sSrc[tid]  = a_src[(size_t)bh * NN + tid];
        sDst[tid]  = a_dst[(size_t)bh * NN + tid];
        sO1[tid]   = o1[(size_t)bh * NN + tid];
        sO2[tid]   = o2[(size_t)bh * NN + tid];
        sDiag[tid] = adj[((size_t)b * NN + tid) * NN + tid];
    }
    __syncthreads();

    if (tid < NN) {
        const float* smr = s_mask + ((size_t)b * NN + tid) * NN;
        const float* adr = adj + ((size_t)b * NN + tid) * NN;
        float as = sSrc[tid], ao = sO1[tid];
        float max2 = -3.0e38f, max3 = -3.0e38f;
        for (int m = 0; m < NN; ++m) {
            float x = as + sDst[m];
            float lg = ((x >= 0.f) ? x : 0.2f * x) + ao + sO2[m];
            bool inter = (smr[m] == 1.0f) && (m != tid);
            bool outer = (smr[m] == 0.0f) && (adr[m] != 0.0f);
            if (inter && lg > max2) max2 = lg;
            if (outer && lg > max3) max3 = lg;
        }
        float sum2 = 0.f, sum3 = 0.f;
        for (int m = 0; m < NN; ++m) {
            float x = as + sDst[m];
            float lg = ((x >= 0.f) ? x : 0.2f * x) + ao + sO2[m];
            bool inter = (smr[m] == 1.0f) && (m != tid);
            bool outer = (smr[m] == 0.0f) && (adr[m] != 0.0f);
            float e2 = inter ? __expf(lg - max2) : 0.f;
            float e3 = outer ? __expf(lg - max3) : 0.f;
            sA2[tid * (NN + 1) + m] = e2; sum2 += e2;
            sA3[tid * (NN + 1) + m] = e3; sum3 += e3;
        }
        float inv2 = (sum2 > 0.f) ? 1.f / sum2 : 0.f;
        float inv3 = (sum3 > 0.f) ? 1.f / sum3 : 0.f;
        for (int m = 0; m < NN; ++m) {
            sA2[tid * (NN + 1) + m] *= inv2;
            sA3[tid * (NN + 1) + m] *= inv3;
        }
    }
    __syncthreads();

    int o = tid;   // 128 threads == OUTD
    float bv = bias[o];
    for (int n = 0; n < NN; ++n) {
        float acc = sDiag[n] * h1[(size_t)n * OUTD + o] + bv;
        for (int m = 0; m < NN; ++m) {
            acc += sA2[n * (NN + 1) + m] * h2s[m * OUTD + o];
            acc += sA3[n * (NN + 1) + m] * h3s[m * OUTD + o];
        }
        acc *= (1.f / 3.f);
        float e = (acc > 0.f) ? acc : (__expf(acc) - 1.f);    // elu
        out[((size_t)b * NN + n) * DD + h * OUTD + o] = e;
    }
}

// ---------------------------------------------------------------- final gated blend (in place on d_out)
__global__ void final_kernel(const float* __restrict__ feat_in, const float* __restrict__ gate_pre,
                             const float* __restrict__ H_b, float* __restrict__ out, int total) {
    int i = blockIdx.x * blockDim.x + threadIdx.x;
    if (i >= total) return;
    int d = i % DD;
    float g = 1.f / (1.f + __expf(-(gate_pre[i] + H_b[d])));
    float fo = out[i];
    out[i] = g * fo + (1.f - g) * feat_in[i];
}

// ---------------------------------------------------------------- host launcher
extern "C" void kernel_launch(void* const* d_in, const int* in_sizes, int n_in,
                              void* d_out, int out_size, void* d_ws, size_t ws_size,
                              hipStream_t stream) {
    (void)in_sizes; (void)n_in; (void)out_size; (void)ws_size;
    const float* feat_in  = (const float*)d_in[0];
    const float* adj      = (const float*)d_in[1];
    const float* s_mask   = (const float*)d_in[2];
    const float* output1  = (const float*)d_in[3];
    const float* output2  = (const float*)d_in[4];
    const float* embedding= (const float*)d_in[5];
    const float* W1       = (const float*)d_in[6];
    const float* W2       = (const float*)d_in[7];
    const float* W3       = (const float*)d_in[8];
    const float* bias     = (const float*)d_in[9];
    const float* w_src    = (const float*)d_in[10];
    const float* w_dst    = (const float*)d_in[11];
    const float* w_src_o  = (const float*)d_in[12];
    const float* w_dst_o  = (const float*)d_in[13];
    const float* H_w      = (const float*)d_in[14];
    const float* H_b      = (const float*)d_in[15];
    float* out = (float*)d_out;

    // workspace carve-up (256B aligned)
    char* ws = (char*)d_ws;
    size_t off = 0;
    auto carve = [&](size_t bytes) -> char* {
        char* p = ws + off;
        off = (off + bytes + 255) & ~(size_t)255;
        return p;
    };
    __bf16* featB   = (__bf16*)carve((size_t)ROWS * DD * 2);              // 29.5 MB
    __bf16* HwB     = (__bf16*)carve((size_t)DD * DD * 2);                //  1.2 MB
    __bf16* WtB     = (__bf16*)carve((size_t)3 * HH * OUTD * DD * 2);     //  3.5 MB
    float*  hbuf    = (float*)carve((size_t)3 * BB * HH * NN * OUTD * 4); // 177 MB
    float*  gatePre = (float*)carve((size_t)ROWS * DD * 4);               // 59 MB
    float*  aSrc    = (float*)carve((size_t)BB * HH * NN * 4);
    float*  aDst    = (float*)carve((size_t)BB * HH * NN * 4);
    float*  o1b     = (float*)carve((size_t)BB * HH * NN * 4);
    float*  o2b     = (float*)carve((size_t)BB * HH * NN * 4);

    // 1) convert feat_in and H_w to bf16
    {
        int n = ROWS * DD;
        cvt_f32_bf16_kernel<<<(n + 255) / 256, 256, 0, stream>>>(feat_in, featB, n);
    }
    {
        int n = DD * DD;
        cvt_f32_bf16_kernel<<<(n + 255) / 256, 256, 0, stream>>>(H_w, HwB, n);
    }
    // 2) transpose+convert weights
    {
        int n = 3 * HH * OUTD * DD;
        cvt_weights_kernel<<<(n + 255) / 256, 256, 0, stream>>>(W1, W2, W3, WtB);
    }
    const size_t gemmLds = 2 * CHUNK_BYTES;   // double-buffered B panel chunks
    // 3) h1/h2/h3 via WMMA bf16 (B panel staged in LDS, shared by 8 waves)
    gemm_h_kernel<<<dim3(MTILES / 8, 3 * HH), 256, gemmLds, stream>>>(featB, WtB, hbuf);
    // 4) gate pre-activation via WMMA bf16
    gemm_gate_kernel<<<dim3(MTILES / 8, DD / 128), 256, gemmLds, stream>>>(featB, HwB, gatePre);
    // 5) argsort + embedding attention terms
    emb_kernel<<<BB, 128, 0, stream>>>(adj, output1, output2, embedding, w_src_o, w_dst_o, o1b, o2b);
    // 6) a_src / a_dst
    {
        int n = BB * HH * NN;
        asrc_kernel<<<(n + 255) / 256, 256, 0, stream>>>(hbuf, w_src, w_dst, aSrc, aDst);
    }
    // 7) masked softmaxes + aggregation + elu -> d_out
    {
        const size_t attnLds = (size_t)(2 * NN * (NN + 1) + 2 * NN * OUTD) * 4;  // 122.4 KB
        attn_kernel<<<BB * HH, 128, attnLds, stream>>>(adj, s_mask, aSrc, aDst, o1b, o2b, hbuf, bias, out);
    }
    // 8) gated blend (in place on d_out)
    {
        int n = ROWS * DD;
        final_kernel<<<(n + 255) / 256, 256, 0, stream>>>(feat_in, gatePre, H_b, out, n);
    }
}
